// Self_Attn_25546465477256
// MI455X (gfx1250) — compile-verified, matrix operand order checked by reading, not run on hardware
//
#include <hip/hip_runtime.h>
#include <hip/hip_bf16.h>

// ---------------------------------------------------------------------------
// Self-attention (B=4, C=256, N=4096, C8=32) for gfx1250 (MI455X, wave32).
//   K0: x (B,C,N) fp32 -> xtb (B,N,C) bf16   (transpose => contiguous K loads)
//   K1: QKV projection via bf16 WMMA; q,k stored transposed (B,N,32) bf16 so
//       energy fragments are two 16B loads; v stored row-major (B,C,N) bf16
//   K2: energy = q^T k (one K=32 WMMA per 16x16 tile) + fused row softmax
//       through 256KB dynamic LDS; fp32 attention -> d_out
//   K3: out = V @ A^T with 32x64 wave tiles (8 chained WMMAs per K-step);
//       epilogue out = gamma*out + x
// ---------------------------------------------------------------------------

typedef __attribute__((ext_vector_type(16))) __bf16 v16bf;
typedef __attribute__((ext_vector_type(8)))  __bf16 v8bf;
typedef __attribute__((ext_vector_type(8)))  float  v8f;
typedef __attribute__((ext_vector_type(4)))  float  v4f;

#define BATCH 4
#define CCH   256
#define C8CH  32
#define NSP   4096   // 64*64 spatial

// 16-bit A/B fragment K-striping (wave32): lane L: m/n=L&15, h=L>>4.
// element e: e<8 -> K = e + 8h ; e>=8 -> K = (e-8) + 16 + 8h
// => per lane the 16 K-values are two contiguous runs of 8: [8h..8h+7] and
//    [16+8h..23+8h]  -> two 16-byte loads when K-axis is contiguous.

// ---------------------------------------------------------------------------
// K0: x (B,C,N) fp32 -> xtb (B,N,C) bf16
// ---------------------------------------------------------------------------
__global__ __launch_bounds__(256) void cvt_tr_kernel(const float* __restrict__ x,
                                                     __bf16* __restrict__ xtb) {
    const size_t i = (size_t)blockIdx.x * 256 + threadIdx.x;  // < B*C*N
    const int n = (int)(i % NSP);
    const size_t bc = i / NSP;
    const int c = (int)(bc % CCH);
    const int b = (int)(bc / CCH);
    xtb[((size_t)b * NSP + n) * CCH + c] = (__bf16)x[i];
}

// ---------------------------------------------------------------------------
// K1: QKV projection. grid = (N/16, 20 rowTiles, B), block = 1 wave.
//     rowTiles 0-1 -> q, 2-3 -> k (stored transposed N x 32),
//     rowTiles 4-19 -> v (stored row-major C x N).
// ---------------------------------------------------------------------------
__global__ __launch_bounds__(32) void proj_kernel(
    const float* __restrict__ Wq, const float* __restrict__ bq,
    const float* __restrict__ Wk, const float* __restrict__ bk,
    const float* __restrict__ Wv, const float* __restrict__ bv,
    const __bf16* __restrict__ xtb,
    __bf16* __restrict__ qtb, __bf16* __restrict__ ktb, __bf16* __restrict__ vb) {

    const int lane = threadIdx.x & 31;
    const int m = lane & 15, h = lane >> 4;
    const int n0 = blockIdx.x * 16;
    const int rt = blockIdx.y;
    const int b  = blockIdx.z;

    const float* W; const float* bias; __bf16* op; int o0; bool trans;
    if (rt < 2)      { W = Wq; bias = bq; op = qtb; o0 = rt * 16;       trans = true;  }
    else if (rt < 4) { W = Wk; bias = bk; op = ktb; o0 = (rt - 2) * 16; trans = true;  }
    else             { W = Wv; bias = bv; op = vb;  o0 = (rt - 4) * 16; trans = false; }

    const float*  Wrow = W   + (size_t)(o0 + m) * CCH;               // A row
    const __bf16* xrow = xtb + ((size_t)b * NSP + n0 + m) * CCH;     // B col

    v8f acc = {};
    for (int k0 = 0; k0 < CCH; k0 += 32) {
        // A: weight rows, fp32 -> bf16 (two contiguous runs of 8)
        const v4f w0 = *(const v4f*)(Wrow + k0 + 8 * h);
        const v4f w1 = *(const v4f*)(Wrow + k0 + 8 * h + 4);
        const v4f w2 = *(const v4f*)(Wrow + k0 + 16 + 8 * h);
        const v4f w3 = *(const v4f*)(Wrow + k0 + 16 + 8 * h + 4);
        // B: x columns, already bf16 and K-contiguous (two 16B loads)
        const v8bf x0 = *(const v8bf*)(xrow + k0 + 8 * h);
        const v8bf x1 = *(const v8bf*)(xrow + k0 + 16 + 8 * h);

        v16bf a, bm;
#pragma unroll
        for (int e = 0; e < 4; ++e) {
            a[e]      = (__bf16)w0[e];
            a[e + 4]  = (__bf16)w1[e];
            a[e + 8]  = (__bf16)w2[e];
            a[e + 12] = (__bf16)w3[e];
        }
#pragma unroll
        for (int e = 0; e < 8; ++e) { bm[e] = x0[e]; bm[e + 8] = x1[e]; }

        acc = __builtin_amdgcn_wmma_f32_16x16x32_bf16(false, a, false, bm,
                                                      (short)0, acc, false, false);
    }

    if (trans) {
        // q/k: store transposed (N x 32); per-lane rows o0+8h..o0+8h+7 are
        // contiguous -> one 16B vector store.
        v8bf st;
#pragma unroll
        for (int r = 0; r < 8; ++r)
            st[r] = (__bf16)(acc[r] + bias[o0 + r + 8 * h]);
        *(v8bf*)(op + ((size_t)b * NSP + n0 + m) * C8CH + o0 + 8 * h) = st;
    } else {
        // v: row-major (C x N)
#pragma unroll
        for (int r = 0; r < 8; ++r) {
            const int o = o0 + r + 8 * h;
            op[((size_t)b * CCH + o) * NSP + n0 + m] = (__bf16)(acc[r] + bias[o]);
        }
    }
}

// ---------------------------------------------------------------------------
// K2: energy + softmax. grid = (N/16, B), block = 256 (8 waves),
//     dynamic LDS = 16*4096*4 = 256 KB. A/B fragments: two v8bf loads each.
// ---------------------------------------------------------------------------
__global__ __launch_bounds__(256) void attn_softmax_kernel(
    const __bf16* __restrict__ qtb, const __bf16* __restrict__ ktb,
    float* __restrict__ attO) {

    extern __shared__ float sE[];  // [16][4096]
    const int tid = threadIdx.x, wave = tid >> 5, lane = tid & 31;
    const int m = lane & 15, h = lane >> 4;
    const int i0 = blockIdx.x * 16, b = blockIdx.y;

    // A: q^T tile, loaded once (K-contiguous per lane).
    const __bf16* qrow = qtb + ((size_t)b * NSP + i0 + m) * C8CH;
    const v8bf q0 = *(const v8bf*)(qrow + 8 * h);
    const v8bf q1 = *(const v8bf*)(qrow + 16 + 8 * h);
    v16bf a;
#pragma unroll
    for (int e = 0; e < 8; ++e) { a[e] = q0[e]; a[e + 8] = q1[e]; }

    for (int jt = wave; jt < NSP / 16; jt += 8) {
        const int j0 = jt * 16;
        const __bf16* krow = ktb + ((size_t)b * NSP + j0 + m) * C8CH;
        const v8bf k0 = *(const v8bf*)(krow + 8 * h);
        const v8bf k1 = *(const v8bf*)(krow + 16 + 8 * h);
        v16bf bf;
#pragma unroll
        for (int e = 0; e < 8; ++e) { bf[e] = k0[e]; bf[e + 8] = k1[e]; }

        v8f acc = {};
        acc = __builtin_amdgcn_wmma_f32_16x16x32_bf16(false, a, false, bf,
                                                      (short)0, acc, false, false);
#pragma unroll
        for (int r = 0; r < 8; ++r)
            sE[(size_t)(r + 8 * h) * NSP + j0 + m] = acc[r];
    }
    __syncthreads();

    // Softmax: wave w owns rows 2w, 2w+1 (full 4096 columns from LDS).
#pragma unroll
    for (int rr = 0; rr < 2; ++rr) {
        const int row = wave * 2 + rr;
        float* rp = sE + (size_t)row * NSP;

        float mx = -3.402823466e38f;
        for (int c = lane; c < NSP; c += 32) mx = fmaxf(mx, rp[c]);
#pragma unroll
        for (int off = 16; off > 0; off >>= 1) mx = fmaxf(mx, __shfl_xor(mx, off, 32));

        float s = 0.f;
        for (int c = lane; c < NSP; c += 32) {
            const float e = __expf(rp[c] - mx);
            rp[c] = e;
            s += e;
        }
#pragma unroll
        for (int off = 16; off > 0; off >>= 1) s += __shfl_xor(s, off, 32);
        const float inv = 1.0f / s;

        float* orow = attO + ((size_t)b * NSP + i0 + row) * NSP;
        for (int c = lane; c < NSP; c += 32) orow[c] = rp[c] * inv;
    }
}

// ---------------------------------------------------------------------------
// K3: out = V @ A^T with 32(c) x 64(i) wave tiles: 2 A frags x 4 B frags ->
//     8 chained K=32 WMMAs per step, 128 steps. grid = (N/64, C/32, B).
//     Epilogue: out = gamma*out + x.
// ---------------------------------------------------------------------------
__global__ __launch_bounds__(32) void av_kernel(
    const __bf16* __restrict__ vb, const float* __restrict__ attO,
    const float* __restrict__ x, const float* __restrict__ gammaP,
    float* __restrict__ outO) {

    const int lane = threadIdx.x & 31;
    const int m = lane & 15, h = lane >> 4;
    const int i0 = blockIdx.x * 64;
    const int c0 = blockIdx.y * 32;
    const int b  = blockIdx.z;

    const __bf16* vrow0 = vb + ((size_t)b * CCH + c0 + m) * NSP;      // A rows, ct=0
    const __bf16* vrow1 = vrow0 + (size_t)16 * NSP;                   // ct=1
    const float*  ar0   = attO + ((size_t)b * NSP + i0 + m) * NSP;    // B cols, it=0

    v8f acc[2][4] = {};

    for (int j0 = 0; j0 < NSP; j0 += 32) {
        v16bf A0, A1;
        {
            const v8bf a00 = *(const v8bf*)(vrow0 + j0 + 8 * h);
            const v8bf a01 = *(const v8bf*)(vrow0 + j0 + 16 + 8 * h);
            const v8bf a10 = *(const v8bf*)(vrow1 + j0 + 8 * h);
            const v8bf a11 = *(const v8bf*)(vrow1 + j0 + 16 + 8 * h);
#pragma unroll
            for (int e = 0; e < 8; ++e) {
                A0[e] = a00[e]; A0[e + 8] = a01[e];
                A1[e] = a10[e]; A1[e + 8] = a11[e];
            }
        }
#pragma unroll
        for (int it = 0; it < 4; ++it) {
            const float* ar = ar0 + (size_t)it * 16 * NSP + j0;
            const v4f b0 = *(const v4f*)(ar + 8 * h);
            const v4f b1 = *(const v4f*)(ar + 8 * h + 4);
            const v4f b2 = *(const v4f*)(ar + 16 + 8 * h);
            const v4f b3 = *(const v4f*)(ar + 16 + 8 * h + 4);
            v16bf bf;
#pragma unroll
            for (int e = 0; e < 4; ++e) {
                bf[e]      = (__bf16)b0[e];
                bf[e + 4]  = (__bf16)b1[e];
                bf[e + 8]  = (__bf16)b2[e];
                bf[e + 12] = (__bf16)b3[e];
            }
            acc[0][it] = __builtin_amdgcn_wmma_f32_16x16x32_bf16(false, A0, false, bf,
                                                                 (short)0, acc[0][it],
                                                                 false, false);
            acc[1][it] = __builtin_amdgcn_wmma_f32_16x16x32_bf16(false, A1, false, bf,
                                                                 (short)0, acc[1][it],
                                                                 false, false);
        }
    }

    const float g = gammaP[0];
#pragma unroll
    for (int ct = 0; ct < 2; ++ct) {
#pragma unroll
        for (int it = 0; it < 4; ++it) {
#pragma unroll
            for (int r = 0; r < 8; ++r) {
                const int c = c0 + ct * 16 + r + 8 * h;
                const size_t idx = ((size_t)b * CCH + c) * NSP + i0 + it * 16 + m;
                outO[idx] = g * acc[ct][it][r] + x[idx];
            }
        }
    }
}

// ---------------------------------------------------------------------------
// Host launcher
// ---------------------------------------------------------------------------
extern "C" void kernel_launch(void* const* d_in, const int* in_sizes, int n_in,
                              void* d_out, int out_size, void* d_ws, size_t ws_size,
                              hipStream_t stream) {
    const float* x     = (const float*)d_in[0];
    const float* Wq    = (const float*)d_in[1];
    const float* bq    = (const float*)d_in[2];
    const float* Wk    = (const float*)d_in[3];
    const float* bk    = (const float*)d_in[4];
    const float* Wv    = (const float*)d_in[5];
    const float* bv    = (const float*)d_in[6];
    const float* gamma = (const float*)d_in[7];

    // d_out = [out (B*C*N fp32) | attention (B*N*N fp32)]
    float* outO = (float*)d_out;
    float* attO = outO + (size_t)BATCH * CCH * NSP;

    // Workspace: xtb (B,N,C) | qtb (B,N,32) | ktb (B,N,32) | vb (B,C,N), bf16
    __bf16* xtb = (__bf16*)d_ws;
    __bf16* qtb = xtb + (size_t)BATCH * NSP * CCH;
    __bf16* ktb = qtb + (size_t)BATCH * NSP * C8CH;
    __bf16* vb  = ktb + (size_t)BATCH * NSP * C8CH;

    // K0: x -> bf16, transposed to (B,N,C)
    {
        const int n = BATCH * CCH * NSP;
        cvt_tr_kernel<<<n / 256, 256, 0, stream>>>(x, xtb);
    }

    // K1: QKV projection
    {
        dim3 grid(NSP / 16, 20, BATCH);   // 2(q) + 2(k) + 16(v) row-tiles
        proj_kernel<<<grid, 32, 0, stream>>>(Wq, bq, Wk, bk, Wv, bv,
                                             xtb, qtb, ktb, vb);
    }

    // K2: energy + softmax (256 KB dynamic LDS per workgroup)
    {
        const size_t lds = (size_t)16 * NSP * sizeof(float);  // 262144 B
        (void)hipFuncSetAttribute((const void*)attn_softmax_kernel,
                                  hipFuncAttributeMaxDynamicSharedMemorySize,
                                  (int)lds);
        dim3 grid(NSP / 16, BATCH);
        attn_softmax_kernel<<<grid, 256, lds, stream>>>(qtb, ktb, attO);
    }

    // K3: out = V @ A^T ; gamma*out + x
    {
        dim3 grid(NSP / 64, CCH / 32, BATCH);
        av_kernel<<<grid, 32, 0, stream>>>(vb, attO, x, gamma, outO);
    }
}
